// PDGSelfAttention_47888885350838
// MI455X (gfx1250) — matrix operand order repaired and checked by reading, not compiled, not run on hardware
//
#include <hip/hip_runtime.h>
#include <hip/hip_bf16.h>
#include <math.h>

typedef __attribute__((ext_vector_type(16))) _Float16 v16h;
typedef __attribute__((ext_vector_type(8)))  _Float16 v8h;
typedef __attribute__((ext_vector_type(4)))  _Float16 v4h;
typedef __attribute__((ext_vector_type(8)))  float    v8f;

constexpr int Bb   = 4;
constexpr int Nn   = 2048;
constexpr int Dd   = 512;
constexpr int ROWS = Bb * Nn;          // 8192
constexpr int LDK  = 40;               // padded k-stride (f16 elems): 80B row pitch -> conflict-free b128
constexpr float SQRT_D = 22.62741699796952f;  // reference divides by d^-0.5

union Frag { v16h v; v8h h[2]; };

// ---- WMMA helpers (CDNA5 wave32, 16x16x32 f16 -> f32) --------------------
__device__ inline v8f wmma16(v16h a, v16h b, v8f c) {
  return __builtin_amdgcn_wmma_f32_16x16x32_f16(
      /*neg_a=*/false, a, /*neg_b=*/false, b,
      /*c_mod=*/(short)0, c, /*reuse_a=*/false, /*reuse_b=*/false);
}

// A fragment (16x32): lane half selects K {0..7,16..23} vs {8..15,24..31}
__device__ inline v16h load_a_frag(const _Float16* lds, int row0, int lane) {
  const int r = lane & 15, half = lane >> 4;
  const _Float16* p = lds + (row0 + r) * LDK + half * 8;
  Frag a;
  a.h[0] = *(const v8h*)p;
  a.h[1] = *(const v8h*)(p + 16);
  return a.v;
}

// B fragment (32x16): lane half selects K {0..7,16..23} vs {8..15,24..31}
// LDS B tile stored [col][k] so each lane reads two runs of 8 contiguous f16 (k and k+16).
__device__ inline v16h load_b_frag(const _Float16* lds, int col0, int lane) {
  const int c = lane & 15, half = lane >> 4;
  const _Float16* p = lds + (col0 + c) * LDK + half * 8;
  Frag b;
  b.h[0] = *(const v8h*)p;
  b.h[1] = *(const v8h*)(p + 16);
  return b.v;
}

__device__ inline float nan2num(float v) {
  if (__builtin_isnan(v)) return 0.0f;
  if (__builtin_isinf(v)) return v > 0.0f ? 1.0f : -1.0f;
  return v;
}

// ---- K0: bias constants: c0 = mean(wd[0,:]), c1 = mean(wd[1,:]), cb = mean(bd)
__global__ __launch_bounds__(256)
void consts_kernel(const float* __restrict__ wd, const float* __restrict__ bd,
                   float* __restrict__ cst) {
  __shared__ float r0[256], r1[256], r2[256];
  const int t = threadIdx.x;
  r0[t] = wd[t]       + wd[t + 256];
  r1[t] = wd[512 + t] + wd[512 + t + 256];
  r2[t] = bd[t]       + bd[t + 256];
  __syncthreads();
  for (int s = 128; s > 0; s >>= 1) {
    if (t < s) { r0[t] += r0[t + s]; r1[t] += r1[t + s]; r2[t] += r2[t + s]; }
    __syncthreads();
  }
  if (t == 0) {
    cst[0] = r0[0] * (1.0f / 512.0f);
    cst[1] = r1[0] * (1.0f / 512.0f);
    cst[2] = r2[0] * (1.0f / 512.0f);
  }
}

// ---- K1: Y[f16] = x[f32] @ W[f32] + b ; tile 128x64, 8 waves (4x2), 32x32/wave
__global__ __launch_bounds__(256)
void qkv_kernel(const float* __restrict__ x, const float* __restrict__ W,
                const float* __restrict__ bias, _Float16* __restrict__ Y) {
  __shared__ _Float16 ldsA[128 * LDK];
  __shared__ _Float16 ldsB[64 * LDK];
  const int tid = threadIdx.x, lane = tid & 31, wv = tid >> 5;
  const int wm = (wv & 3) * 32, wn = (wv >> 2) * 32;
  const int m0 = blockIdx.x * 128;
  const int n0 = blockIdx.y * 64;

  v8f acc[2][2] = {};
  for (int kt = 0; kt < Dd / 32; ++kt) {
    const int k0 = kt * 32;
    __syncthreads();
    { // stage A: x[m0..+127][k0..+31] -> f16
      int r = tid >> 3; const int c = (tid & 7) * 4;
      for (int p = 0; p < 4; ++p, r += 32) {
        float4 f = *(const float4*)(x + (size_t)(m0 + r) * Dd + k0 + c);
        v4h h = {(_Float16)f.x, (_Float16)f.y, (_Float16)f.z, (_Float16)f.w};
        *(v4h*)&ldsA[r * LDK + c] = h;
      }
    }
    { // stage B: W[k0..+31][n0..+63] -> ldsB[n][k] (transpose + f16)
      int k = tid >> 4; const int nq = (tid & 15) * 4;
      for (int p = 0; p < 2; ++p, k += 16) {
        float4 f = *(const float4*)(W + (size_t)(k0 + k) * Dd + n0 + nq);
        ldsB[(nq + 0) * LDK + k] = (_Float16)f.x;
        ldsB[(nq + 1) * LDK + k] = (_Float16)f.y;
        ldsB[(nq + 2) * LDK + k] = (_Float16)f.z;
        ldsB[(nq + 3) * LDK + k] = (_Float16)f.w;
      }
    }
    __syncthreads();
    const v16h a0 = load_a_frag(ldsA, wm,      lane);
    const v16h a1 = load_a_frag(ldsA, wm + 16, lane);
    const v16h b0 = load_b_frag(ldsB, wn,      lane);
    const v16h b1 = load_b_frag(ldsB, wn + 16, lane);
    acc[0][0] = wmma16(a0, b0, acc[0][0]);
    acc[0][1] = wmma16(a0, b1, acc[0][1]);
    acc[1][0] = wmma16(a1, b0, acc[1][0]);
    acc[1][1] = wmma16(a1, b1, acc[1][1]);
  }
  const int cl = lane & 15, half = lane >> 4;
  for (int fm = 0; fm < 2; ++fm)
    for (int fn = 0; fn < 2; ++fn) {
      const int col = n0 + wn + fn * 16 + cl;
      const float bv = bias[col];
      const int rb = m0 + wm + fm * 16 + half * 8;
      for (int e = 0; e < 8; ++e)
        Y[(size_t)(rb + e) * Dd + col] = (_Float16)(acc[fm][fn][e] + bv);
    }
}

// ---- K2: S = sqrt(D)*Q@K^T + (dist . c) + cb, nan_to_num ; tile 128x64
__global__ __launch_bounds__(256)
void scores_kernel(const _Float16* __restrict__ Qh, const _Float16* __restrict__ Kh,
                   const float* __restrict__ dist, const float* __restrict__ cst,
                   float* __restrict__ S) {
  __shared__ _Float16 ldsA[128 * LDK];
  __shared__ _Float16 ldsB[64 * LDK];
  const int tid = threadIdx.x, lane = tid & 31, wv = tid >> 5;
  const int wm = (wv & 3) * 32, wn = (wv >> 2) * 32;
  const int b  = blockIdx.z;
  const int i0 = blockIdx.x * 128;
  const int j0 = blockIdx.y * 64;
  const _Float16* Qb = Qh + (size_t)b * Nn * Dd;
  const _Float16* Kb = Kh + (size_t)b * Nn * Dd;
  const float c0 = cst[0], c1 = cst[1], cb = cst[2];

  v8f acc[2][2] = {};
  for (int kt = 0; kt < Dd / 32; ++kt) {
    const int k0 = kt * 32;
    __syncthreads();
    { // stage A: Q rows i0..+127 (already f16, straight copy)
      int r = tid >> 2; const int cc = (tid & 3) * 8;
      for (int p = 0; p < 2; ++p, r += 64)
        *(v8h*)&ldsA[r * LDK + cc] =
            *(const v8h*)(Qb + (size_t)(i0 + r) * Dd + k0 + cc);
      if (kt + 1 < Dd / 32)
        __builtin_prefetch(Qb + (size_t)(i0 + (tid >> 2)) * Dd + k0 + 32, 0, 0);
    }
    { // stage B: K rows j0..+63 -> ldsB[j][d] (K^T needs K's natural layout)
      const int r = tid >> 2, cc = (tid & 3) * 8;
      *(v8h*)&ldsB[r * LDK + cc] =
          *(const v8h*)(Kb + (size_t)(j0 + r) * Dd + k0 + cc);
    }
    __syncthreads();
    const v16h a0 = load_a_frag(ldsA, wm,      lane);
    const v16h a1 = load_a_frag(ldsA, wm + 16, lane);
    const v16h b0 = load_b_frag(ldsB, wn,      lane);
    const v16h b1 = load_b_frag(ldsB, wn + 16, lane);
    acc[0][0] = wmma16(a0, b0, acc[0][0]);
    acc[0][1] = wmma16(a0, b1, acc[0][1]);
    acc[1][0] = wmma16(a1, b0, acc[1][0]);
    acc[1][1] = wmma16(a1, b1, acc[1][1]);
  }
  const float2* d2 = (const float2*)dist + (size_t)b * Nn * Nn;
  float* Sb = S + (size_t)b * Nn * Nn;
  const int cl = lane & 15, half = lane >> 4;
  for (int fm = 0; fm < 2; ++fm)
    for (int fn = 0; fn < 2; ++fn) {
      const int j  = j0 + wn + fn * 16 + cl;
      const int ib = i0 + wm + fm * 16 + half * 8;
      for (int e = 0; e < 8; ++e) {
        const size_t idx = (size_t)(ib + e) * Nn + j;
        const float2 dd = d2[idx];
        float v = acc[fm][fn][e] * SQRT_D + (dd.x * c0 + dd.y * c1 + cb);
        Sb[idx] = nan2num(v);
      }
    }
}

// ---- K3: in-place row softmax over 2048 cols (one block per row, wave32)
__global__ __launch_bounds__(256)
void softmax_kernel(float* __restrict__ S) {
  __shared__ float red[8];
  const int tid = threadIdx.x, lane = tid & 31, wv = tid >> 5;
  float* row = S + (size_t)blockIdx.x * Nn;
  float v[8];
  float m = -INFINITY;
  for (int i = 0; i < 8; ++i) { v[i] = row[tid + i * 256]; m = fmaxf(m, v[i]); }
  for (int off = 16; off > 0; off >>= 1) m = fmaxf(m, __shfl_xor(m, off));
  if (lane == 0) red[wv] = m;
  __syncthreads();
  m = red[0];
  for (int i = 1; i < 8; ++i) m = fmaxf(m, red[i]);
  float s = 0.0f;
  for (int i = 0; i < 8; ++i) { v[i] = __expf(v[i] - m); s += v[i]; }
  for (int off = 16; off > 0; off >>= 1) s += __shfl_xor(s, off);
  __syncthreads();
  if (lane == 0) red[wv] = s;
  __syncthreads();
  s = 0.0f;
  for (int i = 0; i < 8; ++i) s += red[i];
  const float inv = 1.0f / s;
  for (int i = 0; i < 8; ++i) row[tid + i * 256] = v[i] * inv;
}

// ---- K4: out[f32] = P[f32->f16] @ V[f16] ; tile 128x64 over [N x D], k over N
__global__ __launch_bounds__(256)
void pv_kernel(const float* __restrict__ P, const _Float16* __restrict__ Vh,
               float* __restrict__ out) {
  __shared__ _Float16 ldsA[128 * LDK];
  __shared__ _Float16 ldsB[64 * LDK];
  const int tid = threadIdx.x, lane = tid & 31, wv = tid >> 5;
  const int wm = (wv & 3) * 32, wn = (wv >> 2) * 32;
  const int b  = blockIdx.z;
  const int i0 = blockIdx.x * 128;
  const int d0 = blockIdx.y * 64;
  const float*    Pb = P  + (size_t)b * Nn * Nn;
  const _Float16* Vb = Vh + (size_t)b * Nn * Dd;

  v8f acc[2][2] = {};
  for (int kt = 0; kt < Nn / 32; ++kt) {
    const int k0 = kt * 32;
    __syncthreads();
    { // stage A: P[i0..+127][k0..+31] f32 -> f16
      int r = tid >> 3; const int c = (tid & 7) * 4;
      for (int p = 0; p < 4; ++p, r += 32) {
        float4 f = *(const float4*)(Pb + (size_t)(i0 + r) * Nn + k0 + c);
        v4h h = {(_Float16)f.x, (_Float16)f.y, (_Float16)f.z, (_Float16)f.w};
        *(v4h*)&ldsA[r * LDK + c] = h;
      }
      if (kt + 1 < Nn / 32)
        __builtin_prefetch(Pb + (size_t)(i0 + (tid >> 3)) * Nn + k0 + 32, 0, 0);
    }
    { // stage B: V[k0..+31][d0..+63] -> ldsB[d][m] (transpose, contiguous-K frags)
      const int m = tid >> 3, d8 = (tid & 7) * 8;
      v8h vv = *(const v8h*)(Vb + (size_t)(k0 + m) * Dd + d0 + d8);
      for (int i = 0; i < 8; ++i) ldsB[(d8 + i) * LDK + m] = vv[i];
    }
    __syncthreads();
    const v16h a0 = load_a_frag(ldsA, wm,      lane);
    const v16h a1 = load_a_frag(ldsA, wm + 16, lane);
    const v16h b0 = load_b_frag(ldsB, wn,      lane);
    const v16h b1 = load_b_frag(ldsB, wn + 16, lane);
    acc[0][0] = wmma16(a0, b0, acc[0][0]);
    acc[0][1] = wmma16(a0, b1, acc[0][1]);
    acc[1][0] = wmma16(a1, b0, acc[1][0]);
    acc[1][1] = wmma16(a1, b1, acc[1][1]);
  }
  const int cl = lane & 15, half = lane >> 4;
  for (int fm = 0; fm < 2; ++fm)
    for (int fn = 0; fn < 2; ++fn) {
      const int d  = d0 + wn + fn * 16 + cl;
      const int ib = i0 + wm + fm * 16 + half * 8;
      for (int e = 0; e < 8; ++e)
        out[((size_t)b * Nn + ib + e) * Dd + d] = acc[fm][fn][e];
    }
}

extern "C" void kernel_launch(void* const* d_in, const int* in_sizes, int n_in,
                              void* d_out, int out_size, void* d_ws, size_t ws_size,
                              hipStream_t stream) {
  const float* x    = (const float*)d_in[0];
  const float* dist = (const float*)d_in[1];
  const float* wq   = (const float*)d_in[2];
  const float* bq   = (const float*)d_in[3];
  const float* wk   = (const float*)d_in[4];
  const float* bk   = (const float*)d_in[5];
  const float* wvp  = (const float*)d_in[6];
  const float* bvp  = (const float*)d_in[7];
  const float* wd   = (const float*)d_in[8];
  const float* bd   = (const float*)d_in[9];
  float* out = (float*)d_out;

  // workspace layout: [cst 256B][Qh 8MB][Kh 8MB][Vh 8MB][S 67MB]  (~92.3MB < L2)
  char* ws = (char*)d_ws;
  float*    cst = (float*)ws;
  _Float16* Qh  = (_Float16*)(ws + 256);
  _Float16* Kh  = Qh + (size_t)ROWS * Dd;
  _Float16* Vh  = Kh + (size_t)ROWS * Dd;
  float*    S   = (float*)(ws + 256 + 3 * (size_t)ROWS * Dd * sizeof(_Float16));

  const dim3 blk(256);
  consts_kernel<<<1, blk, 0, stream>>>(wd, bd, cst);
  qkv_kernel<<<dim3(ROWS / 128, Dd / 64), blk, 0, stream>>>(x, wq, bq, Qh);
  qkv_kernel<<<dim3(ROWS / 128, Dd / 64), blk, 0, stream>>>(x, wk, bk, Kh);
  qkv_kernel<<<dim3(ROWS / 128, Dd / 64), blk, 0, stream>>>(x, wvp, bvp, Vh);
  scores_kernel<<<dim3(Nn / 128, Nn / 64, Bb), blk, 0, stream>>>(Qh, Kh, dist, cst, S);
  softmax_kernel<<<dim3(ROWS), blk, 0, stream>>>(S);
  pv_kernel<<<dim3(Nn / 128, Dd / 64, Bb), blk, 0, stream>>>(S, Vh, out);
}
